// GRUModel_74964359184551
// MI455X (gfx1250) — matrix operand order repaired and checked by reading, not compiled, not run on hardware
//
#include <hip/hip_runtime.h>

typedef __attribute__((ext_vector_type(16))) __bf16 v16bf;
typedef __attribute__((ext_vector_type(8)))  float  v8f;

#define HSZ  31
#define ROWS 16          // batch rows per wave (one 16x16 WMMA tile)

// Branch-free fast activations: v_exp_f32 + v_rcp_f32 only.
__device__ __forceinline__ float fast_sigmoid(float a) {
    return __builtin_amdgcn_rcpf(1.f + __expf(-a));
}
__device__ __forceinline__ float fast_tanh(float a) {
    // 1 - 2/(exp(2a)+1); saturates to +/-1 for large |a|, no branches.
    return 1.f - 2.f * __builtin_amdgcn_rcpf(1.f + __expf(a + a));
}

__global__ __launch_bounds__(32)
void gru_fused_wmma(const float* __restrict__ x,
                    const float* __restrict__ Wih,
                    const float* __restrict__ Whh,
                    const float* __restrict__ bih,
                    const float* __restrict__ bhh,
                    const float* __restrict__ Wfc,
                    const float* __restrict__ bfc,
                    float* __restrict__ out,
                    int T)
{
    __shared__ float s_h[ROWS][32];    // f32 hidden state (col 31 = 0), transpose buffer
    __shared__ float s_wfc[2][32];
    __shared__ float s_bfc[2];

    const int lane = threadIdx.x;      // 0..31, wave32
    const int m    = lane & 15;
    const int hi   = lane >> 4;
    const int bg0  = blockIdx.x * ROWS;

    // ---- one-time init: FC weights in LDS, h0 = 0 ----
    for (int i = lane; i < 64; i += 32) {
        const int c = i >> 5, j = i & 31;
        s_wfc[c][j] = (j < HSZ) ? Wfc[c * HSZ + j] : 0.f;
    }
    if (lane < 2) s_bfc[lane] = bfc[lane];
    for (int i = lane; i < ROWS * 32; i += 32) s_h[i >> 5][i & 31] = 0.f;

    // ---- weight B-fragments in GATE-PADDED column layout: col c = 32*g + j ----
    // B layout (32x16): lane n = lane%16, element i -> K = i<8 ? 8*(lane/16)+i : 16+8*(lane/16)+(i-8)
    v16bf Bih[6], Bhh[6];
#pragma unroll
    for (int nt = 0; nt < 6; ++nt) {
        const int c = nt * 16 + m, g = c >> 5, j = c & 31;
        const int row = g * HSZ + j;
#pragma unroll
        for (int i = 0; i < 16; ++i) {
            const int K = (i < 8) ? (8 * hi + i) : (16 + 8 * hi + (i - 8));
            const bool ok = (j < HSZ) && (K < HSZ);
            Bih[nt][i] = (__bf16)(ok ? Wih[row * HSZ + K] : 0.f);
            Bhh[nt][i] = (__bf16)(ok ? Whh[row * HSZ + K] : 0.f);
        }
    }

    // ---- per-lane bias registers for its two hidden columns j1=m, j2=m+16 ----
    const int j1 = m, j2 = m + 16;
    const bool has2 = (j2 < HSZ);                       // lane 15/31 owns padding col
    const float br1  = bih[j1]           + bhh[j1];
    const float bz1  = bih[HSZ + j1]     + bhh[HSZ + j1];
    const float bni1 = bih[2 * HSZ + j1];
    const float bnh1 = bhh[2 * HSZ + j1];
    const float br2  = has2 ? (bih[j2]       + bhh[j2])       : 0.f;
    const float bz2  = has2 ? (bih[HSZ + j2] + bhh[HSZ + j2]) : 0.f;
    const float bni2 = has2 ? bih[2 * HSZ + j2] : 0.f;
    const float bnh2 = has2 ? bhh[2 * HSZ + j2] : 0.f;

    // ---- h carried in registers: lane owns h[M = v+8*hi][j1|j2], v = 0..7 ----
    float h1[8], h2[8];
#pragma unroll
    for (int v = 0; v < 8; ++v) { h1[v] = 0.f; h2[v] = 0.f; }

    // A fragment of h; A layout (16x32): row m = lane%16,
    // element i -> jj=i>>1: K = (jj<4 ? 8*hi+2*jj : 16+8*hi+2*(jj-4)) + (i&1)
    v16bf hA;
#pragma unroll
    for (int i = 0; i < 16; ++i) hA[i] = (__bf16)0.f;

    __syncthreads();

    const float* xrow = x + ((size_t)(bg0 + m) * T) * HSZ;

    for (int t = 0; t < T; ++t) {
        // ---- build x A-fragment straight from global (no LDS staging) ----
        const float* xr = xrow + (size_t)t * HSZ;
        v16bf xA;
#pragma unroll
        for (int i = 0; i < 16; ++i) {
            const int jj = i >> 1;
            const int K  = ((jj < 4) ? (8 * hi + 2 * jj) : (16 + 8 * hi + 2 * (jj - 4))) + (i & 1);
            xA[i] = (__bf16)((K < HSZ) ? xr[K] : 0.f);
        }
        if (t + 1 < T) __builtin_prefetch(xr + HSZ, 0, 1);   // next timestep row

        // ---- 12 WMMAs; r/z tiles chain C so D = x*Wih^T + h*Whh^T directly ----
        v8f accRZ[4];
#pragma unroll
        for (int nt = 0; nt < 4; ++nt) {
            v8f z8 = {0.f, 0.f, 0.f, 0.f, 0.f, 0.f, 0.f, 0.f};
            v8f a  = __builtin_amdgcn_wmma_f32_16x16x32_bf16(false, xA, false, Bih[nt],
                                                             (short)0, z8, false, false);
            accRZ[nt] = __builtin_amdgcn_wmma_f32_16x16x32_bf16(false, hA, false, Bhh[nt],
                                                                (short)0, a, false, false);
        }
        v8f z8 = {0.f, 0.f, 0.f, 0.f, 0.f, 0.f, 0.f, 0.f};
        v8f xn0 = __builtin_amdgcn_wmma_f32_16x16x32_bf16(false, xA, false, Bih[4], (short)0, z8, false, false);
        v8f hn0 = __builtin_amdgcn_wmma_f32_16x16x32_bf16(false, hA, false, Bhh[4], (short)0, z8, false, false);
        v8f xn1 = __builtin_amdgcn_wmma_f32_16x16x32_bf16(false, xA, false, Bih[5], (short)0, z8, false, false);
        v8f hn1 = __builtin_amdgcn_wmma_f32_16x16x32_bf16(false, hA, false, Bhh[5], (short)0, z8, false, false);

        // ---- GRU gates, fully branch-free in registers (C/D: M = v+8*hi, N = m) ----
#pragma unroll
        for (int v = 0; v < 8; ++v) {
            const int M = v + 8 * hi;
            // column j1 = m : r=nt0, z=nt2, n=nt4
            {
                const float r  = fast_sigmoid(accRZ[0][v] + br1);
                const float zz = fast_sigmoid(accRZ[2][v] + bz1);
                const float nn = fast_tanh(xn0[v] + bni1 + r * (hn0[v] + bnh1));
                h1[v] = (1.f - zz) * nn + zz * h1[v];
                s_h[M][j1] = h1[v];
            }
            // column j2 = m+16 : r=nt1, z=nt3, n=nt5 (lane m==15 writes 0 to pad col 31)
            {
                const float r  = fast_sigmoid(accRZ[1][v] + br2);
                const float zz = fast_sigmoid(accRZ[3][v] + bz2);
                const float nn = fast_tanh(xn1[v] + bni2 + r * (hn1[v] + bnh2));
                h2[v] = (1.f - zz) * nn + zz * h2[v];
                s_h[M][j2] = has2 ? h2[v] : 0.f;       // branchless, keeps col31 == 0
            }
        }
        __syncthreads();                                // transpose visible wave-wide

        // ---- rebuild M-major h A-fragment (f32 master -> bf16) ----
#pragma unroll
        for (int i = 0; i < 16; ++i) {
            const int jj = i >> 1;
            const int K  = ((jj < 4) ? (8 * hi + 2 * jj) : (16 + 8 * hi + 2 * (jj - 4))) + (i & 1);
            hA[i] = (__bf16)s_h[m][K];
        }

        // ---- fused FC head: lane -> (row = lane/2, c = lane&1) ----
        {
            const int row = lane >> 1, c = lane & 1;
            float acc = s_bfc[c];
#pragma unroll
            for (int j = 0; j < HSZ; ++j) acc += s_h[row][j] * s_wfc[c][j];
            out[((size_t)(bg0 + row) * T + t) * 2 + c] = acc;
        }
        __syncthreads();                                // protect s_h before next write
    }
}

extern "C" void kernel_launch(void* const* d_in, const int* in_sizes, int n_in,
                              void* d_out, int out_size, void* d_ws, size_t ws_size,
                              hipStream_t stream) {
    const float* x   = (const float*)d_in[0];
    const float* Wih = (const float*)d_in[1];
    const float* Whh = (const float*)d_in[2];
    const float* bih = (const float*)d_in[3];
    const float* bhh = (const float*)d_in[4];
    const float* Wfc = (const float*)d_in[5];
    const float* bfc = (const float*)d_in[6];
    float* out = (float*)d_out;

    const int T = 512;
    const int B = in_sizes[0] / (T * HSZ);      // 2048
    const int nblocks = B / ROWS;               // 128 independent wave-chains

    gru_fused_wmma<<<nblocks, 32, 0, stream>>>(x, Wih, Whh, bih, bhh, Wfc, bfc, out, T);
}
